// EGCL_Multi_73323681677595
// MI455X (gfx1250) — matrix operand order repaired and checked by reading, not compiled, not run on hardware
//
#include <hip/hip_runtime.h>
#include <hip/hip_bf16.h>
#include <math.h>

typedef _Float16 v16h __attribute__((ext_vector_type(16)));
typedef _Float16 v8h  __attribute__((ext_vector_type(8)));
typedef float    v8f  __attribute__((ext_vector_type(8)));
typedef unsigned int u32x4 __attribute__((ext_vector_type(4)));
typedef unsigned int u32x8 __attribute__((ext_vector_type(8)));

#define NN   512      // nodes
#define NHD  8        // heads
#define HDIM 128      // h feature dim
#define MU   256      // mlp units

// ---- LDS layout (bytes) for edge kernel ----
#define ACT_STRIDE 264                     // halves per activation row (padded)
#define W_STRIDE   272                     // halves per weight row (padded)
#define ACT_OFF  0                         // 16 waves * 16 rows * 264 h * 2B = 135168
#define WST_OFF  135168                    // 256 * 272 * 2B = 139264
#define SQN_OFF  274432                    // 16 waves * 16*8 f32 = 8192
#define WE0_OFF  282624                    // 8*256 f32 = 8192
#define EG_OFF   290816                    // 16 waves * 16 f32 = 1024
#define WINF_OFF 291840                    // 16 * 272 * 2B = 8704 (padded Winf^T)
#define WXO_OFF  300544                    // 16 * 272 * 2B = 8704 (padded Wxo^T)
#define PXS_OFF  309248                    // 16 waves * 16*8 f32 = 8192
#define SMEM_BYTES 317440

// ---- workspace layout (bytes) ----
#define WS_AJ     0          // [512][256] f32
#define WS_AI     524288     // [512][256] f32
#define WS_MI     1048576    // [512][256] f32 (accumulator)
#define WS_SHIFT  1572864    // [512][8][3] f32 (accumulator)
#define WS_WE1T   1622016    // [256][256] f16
#define WS_WX0T   1753088    // [256][256] f16
#define WS_WX1T   1884160    // [256][256] f16
#define WS_WXOP   2015232    // [16][256] f16 (Wxo^T zero-padded to 16 rows)
#define WS_WINFP  2023424    // [16][256] f16 (Winf^T zero-padded to 16 rows)

__device__ __forceinline__ float silu_f(float v) {
    return v * __builtin_amdgcn_rcpf(1.0f + __expf(-v));
}
__device__ __forceinline__ float sigmoid_f(float v) {
    return __builtin_amdgcn_rcpf(1.0f + __expf(-v));
}

// --------------------------------------------------------------------------
// TDM: DMA a 256x256 f16 row-major matrix from global into LDS, inserting
// 32B of padding after every 512B row (-> 272-half LDS row stride).
// D# per CDNA5 ISA ch.8: group0 = {count/lds_addr/global_addr/type},
// group1 = {flags+pad, dims, tile dims, strides}; 2D tensor -> groups 2/3 NULL.
__device__ __forceinline__ void tdm_stage_w(const _Float16* __restrict__ src,
                                            unsigned lds_byte_off) {
    unsigned long long ga = (unsigned long long)src;
    u32x4 g0;
    g0[0] = 1u;                                     // count=1, user descriptor
    g0[1] = lds_byte_off;                           // lds_addr (bytes)
    g0[2] = (unsigned)(ga & 0xffffffffULL);         // global_addr[31:0]
    g0[3] = (unsigned)((ga >> 32) & 0x01ffffffULL)  // global_addr[56:32]
          | (2u << 30);                             // type=2 ("image")
    u32x8 g1;
    g1[0] = (1u << 16)      // data_size = 2 bytes
          | (1u << 20)      // pad_enable
          | (6u << 22)      // pad_interval: 128 dwords (=512B) between pads
          | (7u << 25);     // pad_amount: 8 dwords (=32B)
    g1[1] = (256u << 16);   // tensor_dim0[15:0]=256 (hi bits of dword1)
    g1[2] = (256u << 16);   // tensor_dim0[31:16]=0 | tensor_dim1[15:0]=256
    g1[3] = (256u << 16);   // tensor_dim1[31:16]=0 | tile_dim0=256
    g1[4] = 256u;           // tile_dim1=256, tile_dim2=0
    g1[5] = 256u;           // tensor_dim0_stride[31:0]=256 elements
    g1[6] = 0u;
    g1[7] = 0u;
    asm volatile("tensor_load_to_lds %0, %1" :: "s"(g0), "s"(g1) : "memory");
}

// --------------------------------------------------------------------------
__global__ void k_zero(float* __restrict__ p, int n) {
    int i = blockIdx.x * blockDim.x + threadIdx.x;
    if (i < n) p[i] = 0.0f;
}

// --------------------------------------------------------------------------
// transpose + f16-convert per-edge weights; build zero-padded 16-row tails
__global__ void k_wconv(const float* __restrict__ We1, const float* __restrict__ Wx0,
                        const float* __restrict__ Wx1, const float* __restrict__ Wxo,
                        const float* __restrict__ Winf,
                        _Float16* __restrict__ We1T, _Float16* __restrict__ Wx0T,
                        _Float16* __restrict__ Wx1T, _Float16* __restrict__ WxoP,
                        _Float16* __restrict__ WinfP) {
    int idx = blockIdx.x * 256 + threadIdx.x;   // 65536 total
    int n = idx >> 8, k = idx & 255;
    We1T[n * 256 + k] = (_Float16)We1[k * 256 + n];
    Wx0T[n * 256 + k] = (_Float16)Wx0[k * 256 + n];
    Wx1T[n * 256 + k] = (_Float16)Wx1[k * 256 + n];
    if (idx < 4096) {                            // 16 pad rows x 256
        WxoP[idx]  = (n < 8)  ? (_Float16)Wxo[k * 8 + n] : (_Float16)0.0f;
        WinfP[idx] = (n == 0) ? (_Float16)Winf[k]        : (_Float16)0.0f;
    }
}

// --------------------------------------------------------------------------
// per-node: hc = [h, sqh]; aj = hc @ We0[8:200], ai = hc @ We0[200:392] + be0
__global__ void k_nodeprep(const float* __restrict__ x, const float* __restrict__ h,
                           const float* __restrict__ We0, const float* __restrict__ be0,
                           float* __restrict__ aj, float* __restrict__ ai) {
    __shared__ float hc[192];
    int i = blockIdx.x, t = threadIdx.x;
    if (t < 128) hc[t] = h[i * 128 + t];
    if (t < 64) {
        int p = t >> 3, q = t & 7;
        float s = 0.0f;
        #pragma unroll
        for (int d = 0; d < 3; d++) {
            float dx = x[i * 24 + p * 3 + d] - x[i * 24 + q * 3 + d];
            s += dx * dx;
        }
        hc[128 + t] = s;
    }
    __syncthreads();
    float sj = 0.0f, sa = be0[t];
    for (int k = 0; k < 192; k++) {
        float hv = hc[k];
        sj += hv * We0[(8 + k) * 256 + t];
        sa += hv * We0[(200 + k) * 256 + t];
    }
    aj[i * 256 + t] = sj;
    ai[i * 256 + t] = sa;
}

// --------------------------------------------------------------------------
// stage a 16-row padded tail matrix (16x256 -> 16x272 LDS), 32 threads
__device__ __forceinline__ void stage_w16(const _Float16* __restrict__ src,
                                          char* smem, int off, int t) {
    _Float16* dst = (_Float16*)(smem + off);
    int row = t >> 1, seg = t & 1;
    const uint4* s = (const uint4*)(src + row * 256 + seg * 128);
    uint4* d = (uint4*)(dst + row * W_STRIDE + seg * 128);
    #pragma unroll
    for (int q = 0; q < 16; q++) d[q] = s[q];
}

union H16 { v16h v; v8h h[2]; };

// load the wave's 8 A fragments (16 rows x 256 K) from its activation slab
__device__ __forceinline__ void load_afrags(const _Float16* __restrict__ act,
                                            int lane, H16* A) {
    const int row = lane & 15, seg = lane >> 4;
    const _Float16* arow = act + row * ACT_STRIDE;
    #pragma unroll
    for (int ks = 0; ks < 8; ks++) {
        int kb = ks * 32 + seg * 8;
        A[ks].h[0] = *(const v8h*)(arow + kb);
        A[ks].h[1] = *(const v8h*)(arow + kb + 16);
    }
}

// one wave: slab(16x256) @ W^T(LDS) + bias -> silu -> same slab (in-place)
__device__ __forceinline__ void wave_layer(_Float16* __restrict__ act,
                                           const _Float16* __restrict__ wst,
                                           const float* __restrict__ bias, int lane) {
    const int seg = lane >> 4, ncol = lane & 15;
    H16 A[8];
    load_afrags(act, lane, A);
    #pragma unroll
    for (int nt = 0; nt < 16; nt++) {
        float b = bias[nt * 16 + ncol];
        v8f acc = {b, b, b, b, b, b, b, b};
        const _Float16* wbase = wst + (nt * 16 + ncol) * W_STRIDE + seg * 8;
        H16 B[8];
        #pragma unroll
        for (int ks = 0; ks < 8; ks++) {         // preload all B fragments
            const _Float16* wp = wbase + ks * 32;
            B[ks].h[0] = *(const v8h*)(wp);
            B[ks].h[1] = *(const v8h*)(wp + 16);
        }
        #pragma unroll
        for (int ks = 0; ks < 8; ks++)
            acc = __builtin_amdgcn_wmma_f32_16x16x32_f16(
                false, A[ks].v, false, B[ks].v, (short)0, acc, false, false);
        #pragma unroll
        for (int r = 0; r < 8; r++) {
            float v = silu_f(acc[r]);
            act[(seg ? r + 8 : r) * ACT_STRIDE + nt * 16 + ncol] = (_Float16)v;
        }
    }
}

// one wave: slab(16x256) @ padded 16-col tail matrix -> raw f32 accumulator
__device__ __forceinline__ v8f wave_gemm16(const _Float16* __restrict__ act,
                                           const _Float16* __restrict__ w16,
                                           int lane, float binit) {
    const int seg = lane >> 4, ncol = lane & 15;
    H16 A[8];
    load_afrags(act, lane, A);
    v8f acc = {binit, binit, binit, binit, binit, binit, binit, binit};
    const _Float16* wbase = w16 + ncol * W_STRIDE + seg * 8;
    H16 B[8];
    #pragma unroll
    for (int ks = 0; ks < 8; ks++) {
        const _Float16* wp = wbase + ks * 32;
        B[ks].h[0] = *(const v8h*)(wp);
        B[ks].h[1] = *(const v8h*)(wp + 16);
    }
    #pragma unroll
    for (int ks = 0; ks < 8; ks++)
        acc = __builtin_amdgcn_wmma_f32_16x16x32_f16(
            false, A[ks].v, false, B[ks].v, (short)0, acc, false, false);
    return acc;
}

// --------------------------------------------------------------------------
// main edge kernel: block = 16 i-nodes x 16 j-nodes; wave w owns i-node i0+w
__global__ __launch_bounds__(512, 1)
void k_edge(const float* __restrict__ x, const float* __restrict__ aj,
            const float* __restrict__ ai, const float* __restrict__ We0,
            const float* __restrict__ be1, const float* __restrict__ binf,
            const float* __restrict__ bx0, const float* __restrict__ bx1,
            const float* __restrict__ bxo,
            const _Float16* __restrict__ We1T, const _Float16* __restrict__ Wx0T,
            const _Float16* __restrict__ Wx1T, const _Float16* __restrict__ WxoP,
            const _Float16* __restrict__ WinfP,
            float* __restrict__ mi, float* __restrict__ shiftAcc) {
    extern __shared__ __align__(16) char smem[];
    const int tid = threadIdx.x, wave = tid >> 5, lane = tid & 31;
    const int i0 = blockIdx.x * 16, j0 = blockIdx.y * 16;
    const int i = i0 + wave;

    _Float16* act = (_Float16*)(smem + ACT_OFF) + wave * (16 * ACT_STRIDE);
    _Float16* wst = (_Float16*)(smem + WST_OFF);
    float* sqn = (float*)(smem + SQN_OFF) + wave * 128;   // [16 jj][8 h]
    float* we0s = (float*)(smem + WE0_OFF);               // [8][256]
    float* eg = (float*)(smem + EG_OFF) + wave * 16;      // gate per jj
    float* pxS = (float*)(smem + PXS_OFF) + wave * 128;   // [16 jj][8 h]

    // kick off the We1^T weight DMA (overlaps the whole prologue + layer 1)
    if (wave == 0) tdm_stage_w(We1T, WST_OFF);

    // stage We0[0:8][256] (2048 f32) + padded tail matrices
    ((float4*)we0s)[tid] = ((const float4*)We0)[tid];
    if (tid < 32) stage_w16(WinfP, smem, WINF_OFF, tid);
    else if (tid < 64) stage_w16(WxoP, smem, WXO_OFF, tid - 32);

    // per-wave sqn(i, j0+jj, h)
    for (int t = lane; t < 128; t += 32) {
        int jj = t >> 3, hh = t & 7;
        float s = 0.0f;
        #pragma unroll
        for (int d = 0; d < 3; d++) {
            float dx = x[(j0 + jj) * 24 + hh * 3 + d] - x[i * 24 + hh * 3 + d];
            s += dx * dx;
        }
        sqn[jj * 8 + hh] = s;
    }
    __syncthreads();

    // ---- layer 1 (K=8 edge part + hoisted node parts), silu, f16 into slab
    {
        const int row = lane & 15, seg = lane >> 4;
        const int j = j0 + row;
        float sq[8];
        #pragma unroll
        for (int k = 0; k < 8; k++) sq[k] = sqn[row * 8 + k];
        const float* ajr = aj + j * 256;
        const float* air = ai + i * 256;
        for (int c0 = 0; c0 < 128; c0++) {
            int c = seg * 128 + c0;
            float v = air[c] + ajr[c];
            #pragma unroll
            for (int k = 0; k < 8; k++) v += sq[k] * we0s[k * 256 + c];
            act[row * ACT_STRIDE + c] = (_Float16)silu_f(v);
        }
    }

    // ---- layer 2 (phi_e out): m = silu(t1 @ We1 + be1)
    __builtin_amdgcn_s_wait_tensorcnt(0);   // We1^T DMA complete (wave 0)
    __syncthreads();
    wave_layer(act, wst, be1, lane);

    // all waves done reading We1^T -> start Wx0^T DMA, overlap gate + m_i
    __syncthreads();
    if (wave == 0) tdm_stage_w(Wx0T, WST_OFF);

    // ---- gate e = sigmoid(m @ Winf + binf) * off  (WMMA tail pass)
    {
        v8f g = wave_gemm16(act, (const _Float16*)(smem + WINF_OFF), lane, binf[0]);
        if ((lane & 15) == 0) {                  // only column 0 is Winf
            #pragma unroll
            for (int r = 0; r < 8; r++) {
                int jj = (lane >> 4) ? r + 8 : r;
                float e = sigmoid_f(g[r]);
                if (j0 + jj == i) e = 0.0f;      // off-diagonal mask
                eg[jj] = e;
            }
        }
    }

    // ---- m_i partial: sum_j m * e -> global atomic accumulate
    {
        float ev[16];
        #pragma unroll
        for (int jj = 0; jj < 16; jj++) ev[jj] = eg[jj];
        #pragma unroll
        for (int q = 0; q < 8; q++) {
            int c = lane * 8 + q;
            float s = 0.0f;
            #pragma unroll
            for (int jj = 0; jj < 16; jj++) s += (float)act[jj * ACT_STRIDE + c] * ev[jj];
            atomicAdd(&mi[i * 256 + c], s);
        }
    }

    // ---- phi_x hidden layers (in-place slab: m -> t2 -> t3)
    __builtin_amdgcn_s_wait_tensorcnt(0);   // Wx0^T DMA complete (wave 0)
    __syncthreads();
    wave_layer(act, wst, bx0, lane);
    __syncthreads();
    if (wave == 0) tdm_stage_w(Wx1T, WST_OFF);
    __builtin_amdgcn_s_wait_tensorcnt(0);
    __syncthreads();
    wave_layer(act, wst, bx1, lane);

    // ---- px = t3 @ Wxo + bxo (WMMA tail pass), scatter to pxS[jj][h]
    {
        const int seg = lane >> 4, h = lane & 15;
        v8f p = wave_gemm16(act, (const _Float16*)(smem + WXO_OFF), lane,
                            (h < 8) ? bxo[h] : 0.0f);
        if (h < 8) {
            #pragma unroll
            for (int r = 0; r < 8; r++) {
                int jj = seg ? r + 8 : r;
                pxS[jj * 8 + h] = p[r];
            }
        }
    }

    // ---- shift partial = sum_j (dn/norm) * px ; lanes split (jj, h-halves)
    {
        const int jj = lane & 15, seg = lane >> 4;
        const int j = j0 + jj;
        const bool self = (j == i);
        float sh[12];
        #pragma unroll
        for (int q = 0; q < 12; q++) sh[q] = 0.0f;
        if (!self) {
            #pragma unroll
            for (int h2 = 0; h2 < 4; h2++) {
                int hh = seg * 4 + h2;
                float nrm = sqrtf(sqn[jj * 8 + hh] + 1e-8f) + 1.0f;
                float p = pxS[jj * 8 + hh] * __builtin_amdgcn_rcpf(nrm);
                #pragma unroll
                for (int d = 0; d < 3; d++)
                    sh[h2 * 3 + d] =
                        (x[j * 24 + hh * 3 + d] - x[i * 24 + hh * 3 + d]) * p;
            }
        }
        #pragma unroll
        for (int q = 0; q < 12; q++) {
            float v = sh[q];
            v += __shfl_down(v, 8, 16);
            v += __shfl_down(v, 4, 16);
            v += __shfl_down(v, 2, 16);
            v += __shfl_down(v, 1, 16);
            if ((lane & 15) == 0) atomicAdd(&shiftAcc[i * 24 + seg * 12 + q], v);
        }
    }
}

// --------------------------------------------------------------------------
__global__ void k_finx(const float* __restrict__ x, const float* __restrict__ shift,
                       float* __restrict__ xout) {
    int idx = blockIdx.x * blockDim.x + threadIdx.x;
    if (idx < NN * NHD * 3) xout[idx] = x[idx] + shift[idx] * (1.0f / 511.0f);
}

// phi_h: h_new = h + silu(silu([m_i,h]@Wh0+bh0)@Wh1+bh1)@Who + bho
__global__ void k_finh(const float* __restrict__ h, const float* __restrict__ mi,
                       const float* __restrict__ Wh0, const float* __restrict__ bh0,
                       const float* __restrict__ Wh1, const float* __restrict__ bh1,
                       const float* __restrict__ Who, const float* __restrict__ bho,
                       float* __restrict__ hout) {
    __shared__ float t0S[256];
    __shared__ float t1S[256];
    const int i = blockIdx.x, c = threadIdx.x;
    const float* mir = mi + i * 256;
    const float* hr = h + i * 128;
    float s = bh0[c];
    for (int k = 0; k < 256; k++) s += mir[k] * Wh0[k * 256 + c];
    for (int k = 0; k < 128; k++) s += hr[k] * Wh0[(256 + k) * 256 + c];
    t0S[c] = silu_f(s);
    __syncthreads();
    float s2 = bh1[c];
    for (int k = 0; k < 256; k++) s2 += t0S[k] * Wh1[k * 256 + c];
    t1S[c] = silu_f(s2);
    __syncthreads();
    if (c < 128) {
        float s3 = bho[c];
        for (int k = 0; k < 256; k++) s3 += t1S[k] * Who[k * 128 + c];
        hout[i * 128 + c] = hr[c] + s3;
    }
}

// --------------------------------------------------------------------------
extern "C" void kernel_launch(void* const* d_in, const int* in_sizes, int n_in,
                              void* d_out, int out_size, void* d_ws, size_t ws_size,
                              hipStream_t stream) {
    const float* x    = (const float*)d_in[0];
    const float* h    = (const float*)d_in[1];
    const float* We0  = (const float*)d_in[2];
    const float* be0  = (const float*)d_in[3];
    const float* We1  = (const float*)d_in[4];
    const float* be1  = (const float*)d_in[5];
    const float* Winf = (const float*)d_in[6];
    const float* binf = (const float*)d_in[7];
    const float* Wx0  = (const float*)d_in[8];
    const float* bx0  = (const float*)d_in[9];
    const float* Wx1  = (const float*)d_in[10];
    const float* bx1  = (const float*)d_in[11];
    const float* Wxo  = (const float*)d_in[12];
    const float* bxo  = (const float*)d_in[13];
    const float* Wh0  = (const float*)d_in[14];
    const float* bh0  = (const float*)d_in[15];
    const float* Wh1  = (const float*)d_in[16];
    const float* bh1  = (const float*)d_in[17];
    const float* Who  = (const float*)d_in[18];
    const float* bho  = (const float*)d_in[19];

    char* ws = (char*)d_ws;
    float*     aj    = (float*)(ws + WS_AJ);
    float*     ai    = (float*)(ws + WS_AI);
    float*     mi    = (float*)(ws + WS_MI);
    float*     shift = (float*)(ws + WS_SHIFT);
    _Float16*  We1T  = (_Float16*)(ws + WS_WE1T);
    _Float16*  Wx0T  = (_Float16*)(ws + WS_WX0T);
    _Float16*  Wx1T  = (_Float16*)(ws + WS_WX1T);
    _Float16*  WxoP  = (_Float16*)(ws + WS_WXOP);
    _Float16*  WinfP = (_Float16*)(ws + WS_WINFP);

    float* xout = (float*)d_out;              // [512][8][3]
    float* hout = xout + NN * NHD * 3;        // [512][128]

    k_zero<<<(143360 + 255) / 256, 256, 0, stream>>>(mi, 143360);
    k_wconv<<<256, 256, 0, stream>>>(We1, Wx0, Wx1, Wxo, Winf,
                                     We1T, Wx0T, Wx1T, WxoP, WinfP);
    k_nodeprep<<<512, 256, 0, stream>>>(x, h, We0, be0, aj, ai);
    k_edge<<<dim3(32, 32), 512, SMEM_BYTES, stream>>>(
        x, aj, ai, We0, be1, binf, bx0, bx1, bxo,
        We1T, Wx0T, Wx1T, WxoP, WinfP, mi, shift);
    k_finx<<<(NN * NHD * 3 + 255) / 256, 256, 0, stream>>>(x, shift, xout);
    k_finh<<<512, 256, 0, stream>>>(h, mi, Wh0, bh0, Wh1, bh1, Who, bho, hout);
}
